// Q2Linear_72919954752155
// MI455X (gfx1250) — matrix operand order repaired and checked by reading, not compiled, not run on hardware
//
#include <hip/hip_runtime.h>
#include <hip/hip_bf16.h>

typedef __attribute__((ext_vector_type(16))) _Float16 v16h;
typedef __attribute__((ext_vector_type(8)))  _Float16 v8h;
typedef __attribute__((ext_vector_type(4)))  _Float16 v4h;
typedef __attribute__((ext_vector_type(8)))  float    v8f;

#define M_TOK 1024
#define K_IN  4096
#define N_OUT 4096
#define BLK   128
#define NBLK  (K_IN / BLK)   // 32 blocks of 128 along K

// ---------------- prep: x f32 -> f16 (only activation rounding) --------------------------
__global__ __launch_bounds__(256) void q2_cvt_x(const float* __restrict__ x,
                                                _Float16* __restrict__ xh) {
    const size_t i = ((size_t)blockIdx.x * blockDim.x + threadIdx.x) * 4;
    const float4 v = *(const float4*)(x + i);
    v4h h;
    h[0] = (_Float16)v.x; h[1] = (_Float16)v.y;
    h[2] = (_Float16)v.z; h[3] = (_Float16)v.w;
    *(v4h*)(xh + i) = h;
}

// ---------------- prep: w int32 {-2..1} -> f16 (exact conversion) ------------------------
__global__ __launch_bounds__(256) void q2_cvt_w(const int* __restrict__ wq,
                                                _Float16* __restrict__ wh) {
    const size_t i = ((size_t)blockIdx.x * blockDim.x + threadIdx.x) * 4;
    const int4 v = *(const int4*)(wq + i);
    v4h h;
    h[0] = (_Float16)v.x; h[1] = (_Float16)v.y;
    h[2] = (_Float16)v.z; h[3] = (_Float16)v.w;
    *(v4h*)(wh + i) = h;
}

// ---------------- main: block-scaled GEMM on v_wmma_f32_16x16x32_f16 ---------------------
// Each wave computes a 64(M) x 64(N) tile: 4 A-fragments x 4 B-fragments -> 16 C tiles.
// Per K=32 step: 16 b128 loads feed 16 WMMAs (1.0 wmma/load). Per 128-wide K block:
// zero C, 4 chained WMMAs per tile, then acc += C * scale[n, blk] in f32, exactly
// matching the reference's per-block scale-then-sum semantics.
__global__ __launch_bounds__(256) void q2_gemm(const _Float16* __restrict__ xh,
                                               const _Float16* __restrict__ wh,
                                               const float*    __restrict__ scales,
                                               float*          __restrict__ out) {
    const int lane = threadIdx.x & 31;
    const int wave = threadIdx.x >> 5;
    const int wid  = blockIdx.x * 8 + wave;     // 128 blocks * 8 waves = 1024 tiles
    const int mT   = wid & 15;                  // 16 M-tiles of 64 rows
    const int nT   = wid >> 4;                  // 64 N-tiles of 64 cols
    const int l    = lane & 15;
    const int half = lane >> 4;                 // 0: lanes 0-15, 1: lanes 16-31

    const int mBase = mT * 64;
    const int nBase = nT * 64;

    // A fragment rows (M = mBase + i*16 + l), B fragment rows (N = nBase + j*16 + l)
    const _Float16* aRow[4];
    const _Float16* bRow[4];
    aRow[0] = xh + (size_t)(mBase + l) * K_IN;
    bRow[0] = wh + (size_t)(nBase + l) * K_IN;
    #pragma unroll
    for (int t = 1; t < 4; ++t) {
        aRow[t] = aRow[0] + (size_t)(16 * t) * K_IN;
        bRow[t] = bRow[0] + (size_t)(16 * t) * K_IN;
    }

    v8f acc[4][4] = {};

    for (int blk = 0; blk < NBLK; ++blk) {
        // hint next K-block toward the caches while this one computes
        if (blk + 1 < NBLK) {
            __builtin_prefetch(aRow[0] + (blk + 1) * BLK, 0, 3);
            __builtin_prefetch(aRow[2] + (blk + 1) * BLK, 0, 3);
            __builtin_prefetch(bRow[0] + (blk + 1) * BLK, 0, 3);
            __builtin_prefetch(bRow[2] + (blk + 1) * BLK, 0, 3);
        }

        v8f c[4][4] = {};   // fresh f32 accumulator per 128-wide block

        #pragma unroll
        for (int kk = 0; kk < 4; ++kk) {        // 4 x (K=32) WMMA steps per block
            const int k0 = blk * BLK + kk * 32;

            // A 16x32 f16, contiguous k order used here (same as B): lanes 0-15 hold K 0..15, lanes 16-31 hold K 16..31
            v16h a[4];
            #pragma unroll
            for (int i = 0; i < 4; ++i)
                a[i] = *(const v16h*)(aRow[i] + k0 + half * 16);

            // B 32x16 f16, contiguous k order used here (same as A): lanes 0-15 hold K 0..15, lanes 16-31 hold K 16..31
            v16h b[4];
            #pragma unroll
            for (int j = 0; j < 4; ++j)
                b[j] = *(const v16h*)(bRow[j] + k0 + half * 16);

            #pragma unroll
            for (int i = 0; i < 4; ++i)
                #pragma unroll
                for (int j = 0; j < 4; ++j)
                    c[i][j] = __builtin_amdgcn_wmma_f32_16x16x32_f16(
                        /*neg_a=*/false, a[i], /*neg_b=*/false, b[j],
                        /*c_mod=*/(short)0, c[i][j],
                        /*reuse_a=*/false, /*reuse_b=*/false);
        }

        // per-(out_feature, block) scale; each lane owns column N = nBase + j*16 + l
        #pragma unroll
        for (int j = 0; j < 4; ++j) {
            const float s = scales[(size_t)(nBase + j * 16 + l) * NBLK + blk];
            #pragma unroll
            for (int i = 0; i < 4; ++i)
                #pragma unroll
                for (int r = 0; r < 8; ++r)
                    acc[i][j][r] += c[i][j][r] * s;
        }
    }

    // C/D layout: lane (half,l) holds col nBase+j*16+l, rows mBase+i*16+half*8+r
    #pragma unroll
    for (int i = 0; i < 4; ++i) {
        #pragma unroll
        for (int j = 0; j < 4; ++j) {
            const int col = nBase + j * 16 + l;
            #pragma unroll
            for (int r = 0; r < 8; ++r) {
                const int row = mBase + i * 16 + half * 8 + r;
                out[(size_t)row * N_OUT + col] = acc[i][j][r];
            }
        }
    }
}

extern "C" void kernel_launch(void* const* d_in, const int* in_sizes, int n_in,
                              void* d_out, int out_size, void* d_ws, size_t ws_size,
                              hipStream_t stream) {
    const float* x      = (const float*)d_in[0];
    const int*   wq     = (const int*)d_in[1];     // ternary weights as int32
    const float* scales = (const float*)d_in[2];
    float*       out    = (float*)d_out;

    // workspace layout: [x_f16 : 8 MB][w_f16 : 32 MB] — both L2-resident on MI455X
    _Float16* xh = (_Float16*)d_ws;
    _Float16* wh = (_Float16*)((char*)d_ws + (size_t)M_TOK * K_IN * sizeof(_Float16));

    q2_cvt_x<<<(M_TOK * K_IN) / (256 * 4), 256, 0, stream>>>(x, xh);
    q2_cvt_w<<<(N_OUT * K_IN) / (256 * 4), 256, 0, stream>>>(wq, wh);

    // 16 x 64 = 1024 wave-tiles, 8 waves (256 threads) per block -> 128 blocks
    q2_gemm<<<128, 256, 0, stream>>>(xh, wh, scales, out);
}
